// GATConv_69260642615436
// MI455X (gfx1250) — matrix operand order repaired and checked by reading, not compiled, not run on hardware
//
#include <hip/hip_runtime.h>
#include <hip/hip_bf16.h>
#include <math.h>

typedef __attribute__((ext_vector_type(16))) _Float16 v16h;
typedef __attribute__((ext_vector_type(8)))  _Float16 v8h;
typedef __attribute__((ext_vector_type(4)))  _Float16 v4h;
typedef __attribute__((ext_vector_type(8)))  float    v8f;

#define N_NODES 8192
#define FDIM    128
#define ALPHA   0.2f
#define NEGV    (-9e15f)

// ---------------------------------------------------------------------------
// Kernel A: h = x @ W   (fp32)
// ---------------------------------------------------------------------------
__global__ void gat_xw_kernel(const float* __restrict__ x,
                              const float* __restrict__ W,
                              float* __restrict__ h) {
    __shared__ float s_x[16 * FDIM];
    __shared__ float s_W[32 * FDIM];
    const int t  = threadIdx.x;
    const int i0 = blockIdx.x * 16;

#pragma unroll
    for (int c = 0; c < 8; ++c) {
        int idx = c * 256 + t;
        s_x[idx] = x[(size_t)i0 * FDIM + idx];
    }

    const int col  = t & 127;
    const int rsel = t >> 7;
    float acc[8];
#pragma unroll
    for (int it = 0; it < 8; ++it) acc[it] = 0.0f;

    for (int kc = 0; kc < 4; ++kc) {
        __syncthreads();
#pragma unroll
        for (int c = 0; c < 16; ++c) {
            int idx = c * 256 + t;
            s_W[idx] = W[(size_t)(kc * 32) * FDIM + idx];
        }
        __syncthreads();
#pragma unroll
        for (int it = 0; it < 8; ++it) {
            const int rloc = rsel + 2 * it;
            float a = acc[it];
#pragma unroll
            for (int k = 0; k < 32; ++k)
                a = fmaf(s_x[rloc * FDIM + kc * 32 + k], s_W[k * FDIM + col], a);
            acc[it] = a;
        }
    }
#pragma unroll
    for (int it = 0; it < 8; ++it) {
        const int rloc = rsel + 2 * it;
        h[(size_t)(i0 + rloc) * FDIM + col] = acc[it];
    }
}

// ---------------------------------------------------------------------------
// Kernel B: s1 = h @ a1, s2 = h @ a2.  One wave32 per row.
// ---------------------------------------------------------------------------
__global__ void gat_s12_kernel(const float* __restrict__ h,
                               const float* __restrict__ a1,
                               const float* __restrict__ a2,
                               float* __restrict__ s1,
                               float* __restrict__ s2) {
    const int t    = threadIdx.x;
    const int lane = t & 31;
    const int row  = blockIdx.x * 8 + (t >> 5);
    float p1 = 0.0f, p2 = 0.0f;
#pragma unroll
    for (int c = 0; c < 4; ++c) {
        int k = lane + 32 * c;
        float hv = h[(size_t)row * FDIM + k];
        p1 = fmaf(hv, a1[k], p1);
        p2 = fmaf(hv, a2[k], p2);
    }
#pragma unroll
    for (int m = 16; m >= 1; m >>= 1) {
        p1 += __shfl_xor(p1, m);
        p2 += __shfl_xor(p2, m);
    }
    if (lane == 0) { s1[row] = p1; s2[row] = p2; }
}

// ---------------------------------------------------------------------------
// Kernel P: precompute f16 H-panel in this kernel's contiguous-k (k = 16*kh + e) B-register layout.
// Panel layout (per 128-row tile, dense):
//   panel[tile*16384 + ((kc*8+w)*32 + (nl+16*kh))*16 + e]   (halves)
// Value = h[j0+jj][f] with kc=jj>>5, kh=(jj>>4)&1, e=jj&15, w=f>>4, nl=f&15.
// One block per tile (64 blocks).  2 MB total -> L2-resident.
// ---------------------------------------------------------------------------
__global__ void gat_hpanel_kernel(const float* __restrict__ h,
                                  _Float16* __restrict__ panel) {
    const int t  = threadIdx.x;
    const int j0 = blockIdx.x * 128;
    const float4* h4 = reinterpret_cast<const float4*>(h);
    _Float16* pt = panel + (size_t)blockIdx.x * 16384;

#pragma unroll
    for (int pp = 0; pp < 2; ++pp) {
        const int lin = pp * 256 + t;                // 0..511
        const int o   = lin >> 5;                    // K octet (0..15)
        const int f4  = lin & 31;                    // float4 column group
        const int kc  = o >> 2;
        const int kh  = (o >> 1) & 1;
        const int eb  = (o & 1) * 8;
        v8h c0, c1, c2, c3;
#pragma unroll
        for (int q = 0; q < 8; ++q) {
            const int jj = o * 8 + q;
            const float4 hv = h4[(size_t)(j0 + jj) * (FDIM / 4) + f4];
            c0[q] = (_Float16)hv.x;
            c1[q] = (_Float16)hv.y;
            c2[q] = (_Float16)hv.z;
            c3[q] = (_Float16)hv.w;
        }
        const int fb = 4 * f4;
#pragma unroll
        for (int ff = 0; ff < 4; ++ff) {
            const int f  = fb + ff;
            const int w  = f >> 4;
            const int nl = f & 15;
            const int idx = ((kc * 8 + w) * 32 + (nl + 16 * kh)) * 16 + eb;
            const v8h vv = (ff == 0) ? c0 : (ff == 1) ? c1 : (ff == 2) ? c2 : c3;
            *(v8h*)&pt[idx] = vv;
        }
    }
}

// ---------------------------------------------------------------------------
// Kernel C: fused masked-leaky-logits -> online softmax -> P @ H via WMMA.
//  - B fragments arrive via global_load_async_to_lds_b128 from the
//    precomputed panel (overlapped with the softmax VALU work).
//  - A fragments written swizzled (ds_store_b64), read as ds_load_b128 pairs.
//  LDS slot stride 24 halves = 48B: 16B-aligned, bank step 12.
// ---------------------------------------------------------------------------
#define ASTR 776
#define BSTR 776

__global__ void __launch_bounds__(256)
gat_attn_kernel(const float* __restrict__ adj,
                const _Float16* __restrict__ panel,
                const float* __restrict__ s1,
                const float* __restrict__ s2,
                const float* __restrict__ a_bias,
                float* __restrict__ out) {
    __shared__ _Float16 s_pA[4 * ASTR];    // ~6 KB
    __shared__ _Float16 s_hB[32 * BSTR];   // ~48.5 KB
    __shared__ float    s_scale[16];
    __shared__ float    s_l[16];

    const int t     = threadIdx.x;
    const int wave  = t >> 5;
    const int lane  = t & 31;
    const int r     = t >> 4;        // softmax row served (0..15)
    const int rl    = t & 15;        // lane within row group
    const int i0    = blockIdx.x * 16;
    const int khalf = lane >> 4;

    const float s1v  = s1[i0 + r];
    const float bias = a_bias[0];

    float m = -INFINITY;
    float l = 0.0f;
    v8f acc = {};

    const float4* adj4 = reinterpret_cast<const float4*>(adj + (size_t)(i0 + r) * N_NODES);
    const float4* s24  = reinterpret_cast<const float4*>(s2);

    // LDS byte offset of s_hB (low 32 bits of the shared-aperture address)
    const unsigned int hb0 = (unsigned int)(uintptr_t)&s_hB[0];

    for (int j0 = 0; j0 < N_NODES; j0 += 128) {
        // ---- issue async copies of this tile's B-panel (overlaps phase 1a) --
        // 2048 16B chunks per tile; 8 per thread; chunk id -> panel is linear.
        {
            const char* psrc = (const char*)panel + (size_t)(j0 >> 7) * 32768;
#pragma unroll
            for (int c = 0; c < 8; ++c) {
                const int id   = c * 256 + t;          // 0..2047
                const int blk  = id >> 6;              // (kc*8+w)
                const int rem  = id & 63;
                const int ls   = rem >> 1;             // lane slot
                const int half = rem & 1;
                unsigned long long ga =
                    (unsigned long long)(psrc + (size_t)id * 16);
                unsigned int loff =
                    hb0 + (unsigned int)((blk * BSTR + ls * 24 + half * 8) * 2);
                asm volatile("global_load_async_to_lds_b128 %0, %1, off"
                             :: "v"(loff), "v"(ga) : "memory");
            }
        }

        // ---- phase 1a: logits + online softmax bookkeeping ----
        float ev[2][4];
        float tmax = -INFINITY;
#pragma unroll
        for (int c = 0; c < 2; ++c) {
            const int jb = 4 * rl + 64 * c;              // 16B-aligned j base
            const float4 av = adj4[(j0 + jb) >> 2];
            const float4 sv = s24[(j0 + jb) >> 2];
            const float aa[4] = {av.x, av.y, av.z, av.w};
            const float ss[4] = {sv.x, sv.y, sv.z, sv.w};
#pragma unroll
            for (int q = 0; q < 4; ++q) {
                float e = s1v + ss[q] + bias;
                e = (e >= 0.0f) ? e : ALPHA * e;
                e = (aa[q] > 0.0f) ? e : NEGV;
                ev[c][q] = e;
                tmax = fmaxf(tmax, e);
            }
        }
        if (j0 + 128 < N_NODES)
            __builtin_prefetch(&adj4[(j0 + 128 + 4 * rl) >> 2], 0, 1);

#pragma unroll
        for (int mm = 8; mm >= 1; mm >>= 1)
            tmax = fmaxf(tmax, __shfl_xor(tmax, mm));

        const float mnew  = fmaxf(m, tmax);
        const float scale = __expf(m - mnew);
        float psum = 0.0f;
#pragma unroll
        for (int c = 0; c < 2; ++c) {
            v4h pk;
#pragma unroll
            for (int q = 0; q < 4; ++q) {
                const float p = __expf(ev[c][q] - mnew);
                psum += p;
                pk[q] = (_Float16)p;
            }
            // invert this kernel's contiguous-k A-layout: value (row=r, K=j) lives at
            // slot = r + 16*bit4(k), element = k&15, kc = j>>5
            const int j  = 4 * rl + 64 * c;              // base (q=0)
            const int kc = j >> 5;
            const int k  = j & 31;
            const int slot  = r + 16 * ((k >> 4) & 1);
            const int ebase = k & 15;
            *(v4h*)&s_pA[kc * ASTR + slot * 24 + ebase] = pk;
        }
#pragma unroll
        for (int mm = 8; mm >= 1; mm >>= 1)
            psum += __shfl_xor(psum, mm);
        l = l * scale + psum;
        m = mnew;
        if (rl == 0) s_scale[r] = scale;

        // wait for this wave's async B-panel copies, then block-wide sync
        asm volatile("s_wait_asynccnt 0x0" ::: "memory");
        __syncthreads();

        // ---- phase 2: rescale accumulator, 4x WMMA over K=128 ----
#pragma unroll
        for (int rr = 0; rr < 8; ++rr)
            acc[rr] *= s_scale[rr + (khalf << 3)];

#pragma unroll
        for (int kc = 0; kc < 4; ++kc) {
            const int aidx = kc * ASTR + lane * 24;
            const v8h alo = *(const v8h*)&s_pA[aidx];
            const v8h ahi = *(const v8h*)&s_pA[aidx + 8];
            const v16h af = __builtin_shufflevector(alo, ahi,
                0, 1, 2, 3, 4, 5, 6, 7, 8, 9, 10, 11, 12, 13, 14, 15);

            const int bidx = (kc * 8 + wave) * BSTR + lane * 24;
            const v8h blo = *(const v8h*)&s_hB[bidx];
            const v8h bhi = *(const v8h*)&s_hB[bidx + 8];
            const v16h bf = __builtin_shufflevector(blo, bhi,
                0, 1, 2, 3, 4, 5, 6, 7, 8, 9, 10, 11, 12, 13, 14, 15);

            acc = __builtin_amdgcn_wmma_f32_16x16x32_f16(
                false, af, false, bf, (short)0, acc, false, false);
        }
        __syncthreads();
    }

    if (rl == 0) s_l[r] = l;
    __syncthreads();

    const int nloc = lane & 15;
#pragma unroll
    for (int rr = 0; rr < 8; ++rr) {
        const int M = rr + (khalf << 3);
        out[(size_t)(i0 + M) * FDIM + 16 * wave + nloc] = acc[rr] / s_l[M];
    }
}

// ---------------------------------------------------------------------------
// Launcher.  Inputs: x, adj, W, a1, a2, a_bias.  Output: [8192,128] f32.
// Workspace: h (4 MB) + s1/s2 (64 KB) + f16 B-panel (2 MB).
// ---------------------------------------------------------------------------
extern "C" void kernel_launch(void* const* d_in, const int* in_sizes, int n_in,
                              void* d_out, int out_size, void* d_ws, size_t ws_size,
                              hipStream_t stream) {
    const float* x      = (const float*)d_in[0];
    const float* adj    = (const float*)d_in[1];
    const float* W      = (const float*)d_in[2];
    const float* a1     = (const float*)d_in[3];
    const float* a2     = (const float*)d_in[4];
    const float* a_bias = (const float*)d_in[5];
    float* out = (float*)d_out;

    char* ws = (char*)d_ws;
    float*     h     = (float*)ws;                                   // 4 MB
    float*     s1    = (float*)(ws + (size_t)N_NODES * FDIM * 4);    // 32 KB
    float*     s2    = s1 + N_NODES;                                 // 32 KB
    _Float16*  panel = (_Float16*)(ws + (size_t)N_NODES * FDIM * 4
                                      + 2 * (size_t)N_NODES * 4);    // 2 MB

    gat_xw_kernel<<<N_NODES / 16, 256, 0, stream>>>(x, W, h);
    gat_s12_kernel<<<N_NODES / 8, 256, 0, stream>>>(h, a1, a2, s1, s2);
    gat_hpanel_kernel<<<N_NODES / 128, 256, 0, stream>>>(h, panel);
    gat_attn_kernel<<<N_NODES / 16, 256, 0, stream>>>(adj, panel, s1, s2, a_bias, out);
}
